// HierarchicalGNN_74689481277726
// MI455X (gfx1250) — matrix operand order repaired and checked by reading, not compiled, not run on hardware
//
#include <hip/hip_runtime.h>
#include <math.h>

// MI455X / gfx1250, wave32. Dense GEMMs via v_wmma_f32_16x16x32_bf16 (fp32 acc);
// edge gather/scatter phase is the bandwidth roofline (~5-6 GB @ 23.3 TB/s) and
// uses simple streaming kernels with f32 atomics.

typedef __attribute__((ext_vector_type(16))) __bf16 v16bf;
typedef __attribute__((ext_vector_type(8)))  float  v8f;

#define HID   128
#define GNUM  128
#define CSTR  (3*HID)   // combined output row stride (384)

// ---------------- utility ----------------
__global__ void k_fill(float* __restrict__ p, float v, int n) {
  int i = blockIdx.x*blockDim.x + threadIdx.x;
  if (i < n) p[i] = v;
}

__device__ __forceinline__ void atomicMaxF(float* addr, float v) {
  unsigned int* ua = (unsigned int*)addr;
  unsigned int old = *ua;
  while (__uint_as_float(old) < v) {
    unsigned int assumed = old;
    old = atomicCAS(ua, assumed, __float_as_uint(v));
    if (old == assumed) break;
  }
}

// ---------------- WMMA GEMM ----------------
// C[M,N] = A[M,K] @ B[K,N] (+ bias, optional relu). fp32 in/out, bf16 WMMA compute.
// N = NT*16, K % 32 == 0, M % 16 == 0. block = 128 threads = 4 waves; each wave
// owns one 16-row M tile and all NT column tiles. B is staged into LDS as bf16.
template<int NT>
__global__ void __launch_bounds__(128)
k_gemm_wmma(const float* __restrict__ A, const float* __restrict__ B,
            const float* __restrict__ bias, float* __restrict__ C,
            int M, int K, int doRelu) {
  const int N = NT*16;
  extern __shared__ __bf16 Bs[];
  for (int idx = threadIdx.x; idx < K*N; idx += 128) Bs[idx] = (__bf16)B[idx];
  __syncthreads();

  int lane  = threadIdx.x & 31;
  int wave  = threadIdx.x >> 5;
  int mtile = blockIdx.x*4 + wave;
  if (mtile*16 >= M) return;            // wave-uniform: EXEC all-ones for WMMA
  int row0 = mtile*16;

  const float* arow = A + (size_t)(row0 + (lane & 15))*K;
  int kHalf = (lane >> 4) * 8;          // A frag: lanes16-31 shifted by 8 in K
  int bCol  = lane & 15;
  int bK    = (lane >> 4) * 8;          // B frag: same K order as A frag (lanes16-31 shifted by 8)

  v8f acc[NT];
#pragma unroll
  for (int t = 0; t < NT; ++t) {
    v8f z = {0.f,0.f,0.f,0.f,0.f,0.f,0.f,0.f};
    acc[t] = z;
  }

  for (int k0 = 0; k0 < K; k0 += 32) {
    v16bf afrag;
#pragma unroll
    for (int e = 0; e < 16; ++e) {
      // ISA 16-bit A 16x32 layout: elems 0..7 -> K=kHalf+e ; elems 8..15 -> K=kHalf+16+(e-8)
      int kk = (e < 8) ? (kHalf + e) : (kHalf + 8 + e);
      afrag[e] = (__bf16)arow[k0 + kk];
    }
#pragma unroll
    for (int t = 0; t < NT; ++t) {
      v16bf bfrag;
#pragma unroll
      for (int e = 0; e < 16; ++e)
        bfrag[e] = Bs[(k0 + bK + e + (e < 8 ? 0 : 8))*N + t*16 + bCol];
      acc[t] = __builtin_amdgcn_wmma_f32_16x16x32_bf16(
                   false, afrag, false, bfrag, (short)0, acc[t], false, false);
    }
  }

  int rBase = (lane >> 4) * 8;          // f32 C/D layout: VGPR r -> M=r (+8 for hi lanes)
#pragma unroll
  for (int t = 0; t < NT; ++t) {
    int col = t*16 + (lane & 15);
    float bv = bias ? bias[col] : 0.0f;
#pragma unroll
    for (int r = 0; r < 8; ++r) {
      float v = acc[t][r] + bv;
      if (doRelu) v = v > 0.f ? v : 0.f;
      C[(size_t)(row0 + rBase + r)*N + col] = v;
    }
  }
}

// ---------------- GAT attention pieces ----------------
// a_src[n,h] = sum_c H[n,h,c]*att_src[h,c] ; a_dst likewise.
__global__ void k_attn(const float* __restrict__ H, const float* __restrict__ as,
                       const float* __restrict__ ad, float* __restrict__ aS,
                       float* __restrict__ aD, int N, int heads, int C) {
  int i = blockIdx.x*blockDim.x + threadIdx.x;
  if (i >= N*heads) return;
  int n = i / heads, h = i % heads;
  const float* hp = H + (size_t)n*HID + h*C;
  float s = 0.f, d = 0.f;
  for (int c = 0; c < C; ++c) {
    float v = hp[c];
    s = fmaf(v, as[h*C + c], s);
    d = fmaf(v, ad[h*C + c], d);
  }
  aS[i] = s; aD[i] = d;
}

__device__ __forceinline__ void edgeSD(int e, int E, const int* src, const int* dst,
                                       int& s, int& d) {
  if (e < E) { s = src[e]; d = dst[e]; }
  else       { s = e - E; d = e - E; }          // appended self-loops
}

__global__ void k_edge_max(const int* __restrict__ src, const int* __restrict__ dst,
                           const float* __restrict__ aS, const float* __restrict__ aD,
                           float* __restrict__ m, int E, int N, int heads) {
  int i = blockIdx.x*blockDim.x + threadIdx.x;
  if (i >= (E + N)*heads) return;
  int e = i / heads, h = i % heads, s, d;
  edgeSD(e, E, src, dst, s, d);
  float v = aS[s*heads + h] + aD[d*heads + h];
  v = v > 0.f ? v : 0.2f*v;                     // leaky_relu(0.2)
  atomicMaxF(&m[d*heads + h], v);
}

__global__ void k_edge_exp(const int* __restrict__ src, const int* __restrict__ dst,
                           const float* __restrict__ aS, const float* __restrict__ aD,
                           const float* __restrict__ m, float* __restrict__ denom,
                           float* __restrict__ ex, int E, int N, int heads) {
  int i = blockIdx.x*blockDim.x + threadIdx.x;
  if (i >= (E + N)*heads) return;
  int e = i / heads, h = i % heads, s, d;
  edgeSD(e, E, src, dst, s, d);
  float v = aS[s*heads + h] + aD[d*heads + h];
  v = v > 0.f ? v : 0.2f*v;
  float t = __expf(v - m[d*heads + h]);
  ex[i] = t;
  atomicAdd(&denom[d*heads + h], t);
}

// one block (128 threads) per edge: out[dst, :] += H[src, :] * alpha[h]
__global__ void __launch_bounds__(HID)
k_edge_agg(const int* __restrict__ src, const int* __restrict__ dst,
           const float* __restrict__ H, const float* __restrict__ ex,
           const float* __restrict__ denom, float* __restrict__ out,
           int E, int N, int heads, int cshift) {
  int e = blockIdx.x, s, d;
  edgeSD(e, E, src, dst, s, d);
  int c = threadIdx.x;
  int h = c >> cshift;
  float alpha = ex[e*heads + h] / (denom[d*heads + h] + 1e-16f);
  atomicAdd(&out[(size_t)d*HID + c], H[(size_t)s*HID + c] * alpha);
}

// x = elu((x + b)*gamma + beta)   (BatchNorm eval with running stats (0,1))
__global__ void k_bias_bn_elu(float* __restrict__ x, const float* __restrict__ b,
                              const float* __restrict__ gam, const float* __restrict__ bet,
                              int N) {
  int i = blockIdx.x*blockDim.x + threadIdx.x;
  if (i >= N*HID) return;
  int c = i & (HID - 1);
  float v = (x[i] + b[c])*gam[c] + bet[c];
  x[i] = v > 0.f ? v : (__expf(v) - 1.f);
}

__global__ void k_add_bias(float* __restrict__ x, const float* __restrict__ b, int N) {
  int i = blockIdx.x*blockDim.x + threadIdx.x;
  if (i >= N*HID) return;
  x[i] += b[i & (HID - 1)];
}

// ---------------- pooling ----------------
__global__ void k_pool_sum(const float* __restrict__ feat, const int* __restrict__ batch,
                           float* __restrict__ pool, float* __restrict__ cnt, int N) {
  int i = blockIdx.x*blockDim.x + threadIdx.x;
  if (i >= N*HID) return;
  int n = i >> 7, c = i & 127;
  int g = batch[n];
  atomicAdd(&pool[g*HID + c], feat[i]);
  if (c == 0) atomicAdd(&cnt[g], 1.0f);
}

__global__ void k_pool_div(float* __restrict__ pool, const float* __restrict__ cnt,
                           float* __restrict__ combined, int off) {
  int i = blockIdx.x*blockDim.x + threadIdx.x;
  if (i >= GNUM*HID) return;
  int g = i >> 7, c = i & 127;
  float v = pool[i] / fmaxf(cnt[g], 1.0f);
  pool[i] = v;                                  // normalized pool kept for proj head
  combined[g*CSTR + off + c] = v;
}

// ---------------- triplet loss ----------------
__global__ void k_triplet(const float* __restrict__ Z, const int* __restrict__ anchor,
                          const int* __restrict__ neg, float* __restrict__ loss, int nA) {
  __shared__ float red[64];
  int t = threadIdx.x;
  float v = 0.f;
  if (t < nA) {
    int a = anchor[t], p = (a + 1) % GNUM, n = neg[t];
    float dp = 0.f, dn = 0.f;
    for (int k = 0; k < 16; ++k) {
      float x = Z[a*16 + k] - Z[p*16 + k] + 1e-6f; dp += x*x;
      float y = Z[a*16 + k] - Z[n*16 + k] + 1e-6f; dn += y*y;
    }
    v = sqrtf(dp) - sqrtf(dn) + 1.0f;
    v = v > 0.f ? v : 0.f;
  }
  red[t] = v;
  __syncthreads();
  for (int s = 32; s > 0; s >>= 1) { if (t < s) red[t] += red[t + s]; __syncthreads(); }
  if (t == 0) loss[0] = red[0] / (float)nA;
}

// ---------------- launch ----------------
// d_in layout (setup_inputs() dict order; nested dicts in literal order):
//  0 site_x  1 site_ei  2 site_batch  3 route_x 4 route_ei 5 route_batch
//  6 flight_x 7 flight_ei 8 flight_batch
//  9..18 site_params  {W1,as1,ad1,b1,gamma,beta,W2,as2,ad2,b2}
// 19..28 route_params 29..38 flight_params
// 39..42 proj {W1,b1,W2,b2}  43 anchor_idx 44 neg_idx
extern "C" void kernel_launch(void* const* d_in, const int* in_sizes, int n_in,
                              void* d_out, int out_size, void* d_ws, size_t ws_size,
                              hipStream_t stream) {
  (void)n_in; (void)out_size; (void)ws_size;
  const int N  = in_sizes[0] / 32;   // 50000 (multiple of 16)
  const int E  = in_sizes[1] / 2;    // 800000
  const int ET = E + N;              // edges incl. self loops

  float* out = (float*)d_out;
  float* ws  = (float*)d_ws;
  float* bufA  = ws;                          // N*HID  (GEMM outputs H)
  float* bufB  = bufA  + (size_t)N*HID;       // N*HID  (aggregated features)
  float* aSrc  = bufB  + (size_t)N*HID;       // N*4 max
  float* aDst  = aSrc  + (size_t)N*4;
  float* mBuf  = aDst  + (size_t)N*4;
  float* dBuf  = mBuf  + (size_t)N*4;
  float* exBuf = dBuf  + (size_t)N*4;         // ET*4 max
  float* pools = exBuf + (size_t)ET*4;        // 3*GNUM*HID
  float* cnt   = pools + 3*GNUM*HID;          // GNUM
  float* z1    = cnt   + GNUM;                // GNUM*64
  float* z2    = z1    + GNUM*64;             // GNUM*16

  const int pbase[3]    = {9, 19, 29};
  const int headsArr[3] = {4, 2, 2};
  const int xI[3] = {0,3,6}, eI[3] = {1,4,7}, bI[3] = {2,5,8};

  const int gN128      = (N*HID + 255)/256;
  const int gemmBlocks = (N/16 + 3)/4;

  for (int br = 0; br < 3; ++br) {
    const float* x   = (const float*)d_in[xI[br]];
    const int*   ei  = (const int*)  d_in[eI[br]];
    const int*   bat = (const int*)  d_in[bI[br]];
    const int* src = ei; const int* dst = ei + E;
    int pb = pbase[br];
    const float* W1  = (const float*)d_in[pb+0];
    const float* as1 = (const float*)d_in[pb+1];
    const float* ad1 = (const float*)d_in[pb+2];
    const float* b1  = (const float*)d_in[pb+3];
    const float* gam = (const float*)d_in[pb+4];
    const float* bet = (const float*)d_in[pb+5];
    const float* W2  = (const float*)d_in[pb+6];
    const float* as2 = (const float*)d_in[pb+7];
    const float* ad2 = (const float*)d_in[pb+8];
    const float* b2  = (const float*)d_in[pb+9];
    int heads  = headsArr[br];
    int C      = HID / heads;
    int cshift = (C == 32) ? 5 : ((C == 64) ? 6 : 7);

    // ---- GAT layer 1 (multi-head, concat) ----
    k_gemm_wmma<8><<<gemmBlocks, 128, 32*HID*sizeof(__bf16), stream>>>(
        x, W1, nullptr, bufA, N, 32, 0);
    int gNH = (N*heads + 255)/256;
    k_attn<<<gNH, 256, 0, stream>>>(bufA, as1, ad1, aSrc, aDst, N, heads, C);
    k_fill<<<gNH, 256, 0, stream>>>(mBuf, -3.402823466e38f, N*heads);
    k_fill<<<gNH, 256, 0, stream>>>(dBuf, 0.f, N*heads);
    k_fill<<<gN128, 256, 0, stream>>>(bufB, 0.f, N*HID);
    int gEH = (ET*heads + 255)/256;
    k_edge_max<<<gEH, 256, 0, stream>>>(src, dst, aSrc, aDst, mBuf, E, N, heads);
    k_edge_exp<<<gEH, 256, 0, stream>>>(src, dst, aSrc, aDst, mBuf, dBuf, exBuf, E, N, heads);
    k_edge_agg<<<ET, HID, 0, stream>>>(src, dst, bufA, exBuf, dBuf, bufB, E, N, heads, cshift);
    k_bias_bn_elu<<<gN128, 256, 0, stream>>>(bufB, b1, gam, bet, N);

    // ---- GAT layer 2 (1 head) ----
    k_gemm_wmma<8><<<gemmBlocks, 128, HID*HID*sizeof(__bf16), stream>>>(
        bufB, W2, nullptr, bufA, N, HID, 0);
    int gN1 = (N + 255)/256;
    k_attn<<<gN1, 256, 0, stream>>>(bufA, as2, ad2, aSrc, aDst, N, 1, HID);
    k_fill<<<gN1, 256, 0, stream>>>(mBuf, -3.402823466e38f, N);
    k_fill<<<gN1, 256, 0, stream>>>(dBuf, 0.f, N);
    k_fill<<<gN128, 256, 0, stream>>>(bufB, 0.f, N*HID);
    int gE1 = (ET + 255)/256;
    k_edge_max<<<gE1, 256, 0, stream>>>(src, dst, aSrc, aDst, mBuf, E, N, 1);
    k_edge_exp<<<gE1, 256, 0, stream>>>(src, dst, aSrc, aDst, mBuf, dBuf, exBuf, E, N, 1);
    k_edge_agg<<<ET, HID, 0, stream>>>(src, dst, bufA, exBuf, dBuf, bufB, E, N, 1, 7);
    k_add_bias<<<gN128, 256, 0, stream>>>(bufB, b2, N);

    // ---- mean pool into combined[:, br*HID : (br+1)*HID] ----
    float* poolB = pools + br*GNUM*HID;
    k_fill<<<(GNUM*HID + 255)/256, 256, 0, stream>>>(poolB, 0.f, GNUM*HID);
    k_fill<<<1, 256, 0, stream>>>(cnt, 0.f, GNUM);
    k_pool_sum<<<gN128, 256, 0, stream>>>(bufB, bat, poolB, cnt, N);
    k_pool_div<<<(GNUM*HID + 255)/256, 256, 0, stream>>>(poolB, cnt, out, br*HID);
  }

  // ---- projection head on site pool + triplet loss ----
  const float* Wp1 = (const float*)d_in[39];
  const float* bp1 = (const float*)d_in[40];
  const float* Wp2 = (const float*)d_in[41];
  const float* bp2 = (const float*)d_in[42];
  const int* anchor = (const int*)d_in[43];
  const int* negidx = (const int*)d_in[44];
  int nA = in_sizes[43];

  int pBlocks = (GNUM/16 + 3)/4;
  k_gemm_wmma<4><<<pBlocks, 128, HID*64*sizeof(__bf16), stream>>>(
      pools, Wp1, bp1, z1, GNUM, HID, 1);           // relu(pool@W1 + b1) -> [G,64]
  k_gemm_wmma<1><<<pBlocks, 128, 64*16*sizeof(__bf16), stream>>>(
      z1, Wp2, bp2, z2, GNUM, 64, 0);               // z1@W2 + b2 -> [G,16]
  k_triplet<<<1, 64, 0, stream>>>(z2, anchor, negidx, out + GNUM*CSTR, nA);
}
